// Time_att_48885317763143
// MI455X (gfx1250) — matrix operand order repaired and checked by reading, not compiled, not run on hardware
//
#include <hip/hip_runtime.h>
#include <hip/hip_bf16.h>

typedef __attribute__((ext_vector_type(16))) __bf16 v16bf;
typedef __attribute__((ext_vector_type(8)))  float  v8f;

union Frag  { v16bf v; uint4 u[2]; };
union Pack8 { __bf16 h[8]; uint4 u; };

#define B_  16
#define L_  2048
#define D_  512

__device__ __forceinline__ __bf16 f2bf(float f) { return (__bf16)f; }

// ---------------- fp32 -> bf16 conversion of x ----------------
__global__ void k_cvt_x(const float* __restrict__ x, __bf16* __restrict__ xb, int n) {
  int i = (blockIdx.x * blockDim.x + threadIdx.x) * 4;
  if (i >= n) return;
  float4 f = *(const float4*)(x + i);
  union { __bf16 h[4]; uint2 u; } p;
  p.h[0] = f2bf(f.x); p.h[1] = f2bf(f.y); p.h[2] = f2bf(f.z); p.h[3] = f2bf(f.w);
  *(uint2*)(xb + i) = p.u;
}

// ------------- transpose + convert the three weights -------------
__global__ void k_cvt_w(const float* __restrict__ Wq, const float* __restrict__ Wk,
                        const float* __restrict__ Wv,
                        __bf16* __restrict__ Wtq, __bf16* __restrict__ Wtk,
                        __bf16* __restrict__ Wtv) {
  int idx = blockIdx.x * blockDim.x + threadIdx.x;   // 3 * 512 * 512 total
  int sel = idx >> 18;
  int rem = idx & 262143;
  int n = rem >> 9, k = rem & 511;
  const float* W  = (sel == 0) ? Wq : (sel == 1) ? Wk : Wv;
  __bf16*      Wt = (sel == 0) ? Wtq : (sel == 1) ? Wtk : Wtv;
  Wt[(n << 9) + k] = f2bf(W[(k << 9) + n]);
}

// ---------------- QKV projection GEMM (WMMA bf16) ----------------
// blockIdx.z: 0 -> Q (scaled 1/8, row-major), 1 -> K (row-major), 2 -> V (stored transposed Vt[b][d][l])
__global__ __launch_bounds__(256) void k_gemm_qkv(
    const __bf16* __restrict__ xb,
    const __bf16* __restrict__ Wtq, const __bf16* __restrict__ Wtk, const __bf16* __restrict__ Wtv,
    const float* __restrict__ bq, const float* __restrict__ bk, const float* __restrict__ bv,
    __bf16* __restrict__ Q, __bf16* __restrict__ K, __bf16* __restrict__ Vt)
{
  const int lane = threadIdx.x & 31, wave = threadIdx.x >> 5;
  const int nl = lane & 15, half = lane >> 4;
  const int wm = wave & 3, wn = wave >> 2;
  const int mbase = blockIdx.x * 128 + wm * 32;
  const int nbase = blockIdx.y * 64 + wn * 32;
  const int sel = blockIdx.z;
  const __bf16* Wt   = (sel == 0) ? Wtq : (sel == 1) ? Wtk : Wtv;
  const float*  bias = (sel == 0) ? bq  : (sel == 1) ? bk  : bv;

  v8f acc[2][2] = {};
  for (int ks = 0; ks < 16; ++ks) {
    const int kb = ks * 32;
    Frag a[2], b[2];
#pragma unroll
    for (int mt = 0; mt < 2; ++mt) {
      const __bf16* p = xb + (size_t)(mbase + mt * 16 + nl) * D_ + kb + half * 8;
      a[mt].u[0] = *(const uint4*)p;
      a[mt].u[1] = *(const uint4*)(p + 16);
    }
#pragma unroll
    for (int nt = 0; nt < 2; ++nt) {
      const __bf16* p = Wt + (size_t)(nbase + nt * 16 + nl) * D_ + kb + half * 8;
      b[nt].u[0] = *(const uint4*)p;
      b[nt].u[1] = *(const uint4*)(p + 16);
    }
#pragma unroll
    for (int mt = 0; mt < 2; ++mt)
#pragma unroll
      for (int nt = 0; nt < 2; ++nt)
        acc[mt][nt] = __builtin_amdgcn_wmma_f32_16x16x32_bf16(
            false, a[mt].v, false, b[nt].v, (short)0, acc[mt][nt], false, false);
  }

#pragma unroll
  for (int mt = 0; mt < 2; ++mt) {
    const int gm0 = mbase + mt * 16 + half * 8;   // 8 consecutive M rows per lane
#pragma unroll
    for (int nt = 0; nt < 2; ++nt) {
      const int n = nbase + nt * 16 + nl;
      const float bv_ = bias[n];
      if (sel == 2) {
        Pack8 pk;
#pragma unroll
        for (int r = 0; r < 8; ++r) pk.h[r] = f2bf(acc[mt][nt][r] + bv_);
        const int batch = gm0 >> 11, pos = gm0 & (L_ - 1);
        *(uint4*)(Vt + ((size_t)(batch * D_ + n) << 11) + pos) = pk.u;
      } else {
        __bf16* outp = (sel == 0) ? Q : K;
        const float sc = (sel == 0) ? 0.125f : 1.0f;   // 1/kd = 1/sqrt(512/8)
#pragma unroll
        for (int r = 0; r < 8; ++r)
          outp[(size_t)(gm0 + r) * D_ + n] = f2bf((acc[mt][nt][r] + bv_) * sc);
      }
    }
  }
}

// ------------- flash attention + residual + LayerNorm -------------
__global__ __launch_bounds__(256) void k_attn(
    const __bf16* __restrict__ Q, const __bf16* __restrict__ K, const __bf16* __restrict__ Vt,
    const float* __restrict__ x, const float* __restrict__ w1, const float* __restrict__ b1,
    const float* __restrict__ gamma, const float* __restrict__ beta,
    float* __restrict__ out)
{
  __shared__ __align__(16) __bf16 Qs[16 * D_];      // 16 KB
  __shared__ __align__(16) float  Sbuf[16 * 128];   // 8 KB
  __shared__ __align__(16) __bf16 Pbuf[16 * 128];   // 4 KB
  __shared__ __align__(16) float  Ybuf[16 * D_];    // 32 KB
  __shared__ float m_arr[16], l_arr[16], alpha_arr[16];

  const int tid = threadIdx.x;
  const int lane = tid & 31, wave = tid >> 5;
  const int nl = lane & 15, half = lane >> 4;
  const int batch = blockIdx.x >> 7;           // L/16 = 128 tiles per batch
  const int qrow0 = (blockIdx.x & 127) * 16;

  { // stage 16x512 bf16 Q tile (contiguous 16 KB)
    const uint4* src = (const uint4*)(Q + (size_t)(batch * L_ + qrow0) * D_);
    uint4* dst = (uint4*)Qs;
    for (int i = tid; i < 1024; i += 256) dst[i] = src[i];
  }
  if (tid < 16) { m_arr[tid] = -INFINITY; l_arr[tid] = 0.f; }
  __syncthreads();

  v8f acc[4] = {};     // wave owns output columns [wave*64, wave*64+64)

  for (int kb = 0; kb < 16; ++kb) {
    const int key0 = kb * 128;
    const __bf16* Kb = K + (size_t)(batch * L_ + key0 + wave * 16 + nl) * D_;
    if (kb + 1 < 16)   // prefetch next K block (global_prefetch_b8)
      __builtin_prefetch(K + (size_t)(batch * L_ + key0 + 128 + wave * 16 + nl) * D_, 0, 1);

    // ---- S tile: 16x16 per wave, reduce over d = 512 ----
    v8f s = {};
#pragma unroll
    for (int ks = 0; ks < 16; ++ks) {
      Frag a, b;
      const __bf16* qp = Qs + nl * D_ + ks * 32 + half * 8;
      a.u[0] = *(const uint4*)qp;
      a.u[1] = *(const uint4*)(qp + 16);
      const __bf16* kp = Kb + ks * 32 + half * 8;
      b.u[0] = *(const uint4*)kp;
      b.u[1] = *(const uint4*)(kp + 16);
      s = __builtin_amdgcn_wmma_f32_16x16x32_bf16(false, a.v, false, b.v, (short)0, s, false, false);
    }
#pragma unroll
    for (int r = 0; r < 8; ++r) Sbuf[(r + half * 8) * 128 + wave * 16 + nl] = s[r];
    __syncthreads();

    // ---- online softmax: wave handles rows 2w, 2w+1 ----
#pragma unroll
    for (int rr = 0; rr < 2; ++rr) {
      const int row = wave * 2 + rr;
      float sv[4], rmax = -INFINITY;
#pragma unroll
      for (int j = 0; j < 4; ++j) { sv[j] = Sbuf[row * 128 + lane + j * 32]; rmax = fmaxf(rmax, sv[j]); }
#pragma unroll
      for (int off = 16; off > 0; off >>= 1) rmax = fmaxf(rmax, __shfl_xor(rmax, off, 32));
      const float m_old = m_arr[row];
      const float m_new = fmaxf(m_old, rmax);
      const float alpha = __expf(m_old - m_new);
      float psum = 0.f;
#pragma unroll
      for (int j = 0; j < 4; ++j) {
        const float p = __expf(sv[j] - m_new);
        psum += p;
        Pbuf[row * 128 + lane + j * 32] = f2bf(p);
      }
#pragma unroll
      for (int off = 16; off > 0; off >>= 1) psum += __shfl_xor(psum, off, 32);
      if (lane == 0) {
        m_arr[row] = m_new;
        l_arr[row] = l_arr[row] * alpha + psum;
        alpha_arr[row] = alpha;
      }
    }
    __syncthreads();

    // ---- rescale accumulators by alpha[row] ----
#pragma unroll
    for (int r = 0; r < 8; ++r) {
      const float a_ = alpha_arr[r + half * 8];
#pragma unroll
      for (int nt = 0; nt < 4; ++nt) acc[nt][r] *= a_;
    }

    // ---- O += P @ V  (B-frags straight from transposed Vt) ----
    const __bf16* Vb = Vt + ((size_t)(batch * D_ + wave * 64) << 11) + key0;
#pragma unroll
    for (int kt = 0; kt < 4; ++kt) {
      Frag a;
      const __bf16* pp = Pbuf + nl * 128 + kt * 32 + half * 8;
      a.u[0] = *(const uint4*)pp;
      a.u[1] = *(const uint4*)(pp + 16);
#pragma unroll
      for (int nt = 0; nt < 4; ++nt) {
        Frag b;
        const __bf16* vp = Vb + ((size_t)(nt * 16 + nl) << 11) + kt * 32 + half * 8;
        b.u[0] = *(const uint4*)vp;
        b.u[1] = *(const uint4*)(vp + 16);
        acc[nt] = __builtin_amdgcn_wmma_f32_16x16x32_bf16(false, a.v, false, b.v, (short)0, acc[nt], false, false);
      }
    }
  }

  // ---- epilogue: normalize, residual, *sum(w1)+b1, stage y ----
  float sw = 0.f;
#pragma unroll
  for (int i = 0; i < 8; ++i) sw += w1[i];
  const float b1v = b1[0];

  float invl[8];
#pragma unroll
  for (int r = 0; r < 8; ++r) invl[r] = 1.f / l_arr[r + half * 8];

#pragma unroll
  for (int nt = 0; nt < 4; ++nt) {
    const int n = wave * 64 + nt * 16 + nl;
#pragma unroll
    for (int r = 0; r < 8; ++r) {
      const int m = r + half * 8;
      const float o = acc[nt][r] * invl[r];
      const float xv = x[(size_t)(batch * L_ + qrow0 + m) * D_ + n];
      Ybuf[m * D_ + n] = xv + o * sw + b1v;
    }
  }
  __syncthreads();

  // ---- LayerNorm over D=512: wave handles rows 2w, 2w+1 ----
#pragma unroll
  for (int rr = 0; rr < 2; ++rr) {
    const int row = wave * 2 + rr;
    float sum = 0.f, sq = 0.f;
#pragma unroll
    for (int k = 0; k < 16; ++k) {
      const float v = Ybuf[row * D_ + lane + k * 32];
      sum += v; sq += v * v;
    }
#pragma unroll
    for (int off = 16; off > 0; off >>= 1) { sum += __shfl_xor(sum, off, 32); sq += __shfl_xor(sq, off, 32); }
    const float mean = sum * (1.f / 512.f);
    const float var  = sq  * (1.f / 512.f) - mean * mean;
    const float rs   = rsqrtf(var + 1e-5f);
    float* op = out + (size_t)(batch * L_ + qrow0 + row) * D_;
#pragma unroll
    for (int k = 0; k < 16; ++k) {
      const int col = lane + k * 32;
      op[col] = (Ybuf[row * D_ + col] - mean) * rs * gamma[col] + beta[col];
    }
  }
}

extern "C" void kernel_launch(void* const* d_in, const int* in_sizes, int n_in,
                              void* d_out, int out_size, void* d_ws, size_t ws_size,
                              hipStream_t stream) {
  const float* x     = (const float*)d_in[0];
  const float* Wq    = (const float*)d_in[1];
  const float* bq    = (const float*)d_in[2];
  const float* Wk    = (const float*)d_in[3];
  const float* bk    = (const float*)d_in[4];
  const float* Wv    = (const float*)d_in[5];
  const float* bv    = (const float*)d_in[6];
  const float* w1    = (const float*)d_in[7];
  const float* b1    = (const float*)d_in[8];
  const float* gamma = (const float*)d_in[9];
  const float* beta  = (const float*)d_in[10];
  float* out = (float*)d_out;

  const size_t NX = (size_t)B_ * L_ * D_;          // 16,777,216 elements
  char* ws = (char*)d_ws;
  __bf16* xb  = (__bf16*)ws;  ws += NX * 2;
  __bf16* wtq = (__bf16*)ws;  ws += (size_t)D_ * D_ * 2;
  __bf16* wtk = (__bf16*)ws;  ws += (size_t)D_ * D_ * 2;
  __bf16* wtv = (__bf16*)ws;  ws += (size_t)D_ * D_ * 2;
  __bf16* Qb  = (__bf16*)ws;  ws += NX * 2;
  __bf16* Kb  = (__bf16*)ws;  ws += NX * 2;
  __bf16* Vtb = (__bf16*)ws;  ws += NX * 2;

  k_cvt_x<<<(int)(NX / 1024), 256, 0, stream>>>(x, xb, (int)NX);
  k_cvt_w<<<3 * D_ * D_ / 256, 256, 0, stream>>>(Wq, Wk, Wv, wtq, wtk, wtv);
  k_gemm_qkv<<<dim3((B_ * L_) / 128, D_ / 64, 3), 256, 0, stream>>>(
      xb, wtq, wtk, wtv, bq, bk, bv, Qb, Kb, Vtb);
  k_attn<<<B_ * (L_ / 16), 256, 0, stream>>>(Qb, Kb, Vtb, x, w1, b1, gamma, beta, out);
}
